// VectorQuantizer_13477607375677
// MI455X (gfx1250) — compile-verified, matrix operand order checked by reading, not run on hardware
//
#include <hip/hip_runtime.h>
#include <stdint.h>

// ---------------- problem constants ----------------
#define K_CB   1024
#define D_DIM  256
#define N_PIX  32768
#define N_ELEM 8388608        // 32*256*32*32
#define CT_TILES 64           // K_CB / 16
#define DT_TILES 8            // D_DIM / 32

// ---------------- workspace layout (bytes) ----------------
// Packed codebook: per column tile ct, 16 KB block = [0..2047] bf16-hi u32s,
// [2048..4095] bf16-lo u32s (8 d-subtiles x 256 u32 each). 64 tiles = 1 MB.
#define WS_CBPK 0
#define WS_NRM  (1024*1024)               // 1024 f32 codeword norms
#define WS_IDX  (WS_NRM + 4*1024)         // 32768 int indices
#define WS_PART (WS_IDX + 128*1024)       // 32768 f32 block partial sums

// ---------------- output layout (f32 elements) ----------------
#define OUT_Q    1
#define OUT_ENC  (1 + N_ELEM)             // 8388609
#define OUT_IDX  (OUT_ENC + 33554432)     // 41943041

typedef __attribute__((ext_vector_type(16))) __bf16 v16bf;
typedef __attribute__((ext_vector_type(8)))  float  v8f;
typedef uint32_t u32x4 __attribute__((ext_vector_type(4)));
typedef int      i32x4 __attribute__((ext_vector_type(4)));
typedef int      i32x8 __attribute__((ext_vector_type(8)));

union U16bf { uint4 q[2]; v16bf v; };

#if defined(__has_builtin)
#if __has_builtin(__builtin_amdgcn_tensor_load_to_lds) && __has_builtin(__builtin_amdgcn_s_wait_tensorcnt)
#define HAVE_TDM 1
#endif
#endif
#ifndef HAVE_TDM
#define HAVE_TDM 0
#endif

__device__ __forceinline__ uint32_t bf16_rn_bits(float f) {
    uint32_t u = __float_as_uint(f);
    u += 0x7FFFu + ((u >> 16) & 1u);      // round-to-nearest-even
    return u >> 16;
}
__device__ __forceinline__ float bf16_bits_to_f(uint32_t h) {
    return __uint_as_float(h << 16);
}

#if HAVE_TDM
// One 1-D TDM descriptor: copy 4096 x 4B (16 KB) global -> LDS.
// D# group0: count=1 | lds_addr | global_addr(57b) | type=2 (bits 127:126)
// D# group1: data_size=2(4B)@[17:16]; tensor_dim0=4096@[79:48]; tensor_dim1=1@[111:80];
//            tile_dim0=4096@[127:112]; tensor_dim0_stride=4096@[207:160]
__device__ __forceinline__ void tdm_load_16kb(const uint32_t* gsrc, uint32_t lds_off) {
    uint64_t ga = (uint64_t)(uintptr_t)gsrc;
    u32x4 g0;
    g0[0] = 1u;                                                  // count=1
    g0[1] = lds_off;                                             // LDS byte address
    g0[2] = (uint32_t)ga;                                        // global_addr[31:0]
    g0[3] = (uint32_t)((ga >> 32) & 0x01FFFFFFu) | 0x80000000u;  // ga[56:32] | type=2
    i32x8 g1;
    g1[0] = 0x00020000;            // workgroup_mask=0 (not in cluster), data_size=4B
    g1[1] = (int)(4096u << 16);    // tensor_dim0[15:0] << 16 (abar addr = 0)
    g1[2] = 0x00010000;            // tensor_dim0[31:16]=0 | tensor_dim1=1
    g1[3] = (int)(4096u << 16);    // tensor_dim1[31:16]=0 | tile_dim0=4096
    g1[4] = 0;                     // tile_dim1=0, tile_dim2=0 (1-D)
    g1[5] = 4096;                  // tensor_dim0_stride[31:0]
    g1[6] = 0;
    g1[7] = 0;
    i32x4 z4 = {0, 0, 0, 0};
#if __clang_major__ >= 23
    i32x8 z8 = {0, 0, 0, 0, 0, 0, 0, 0};
    __builtin_amdgcn_tensor_load_to_lds(g0, g1, z4, z4, z8, 0);
#else
    __builtin_amdgcn_tensor_load_to_lds(g0, g1, z4, z4, 0);
#endif
}
#endif

// ============================================================
// Kernel 1: pack codebook into WMMA B-operand register layout.
// Tile (ct,dt) = 32x16 bf16 (K x N); lane holds 8 u32 (v16bf) so a wave
// loads a tile with two ds_load_b128 per lane. hi block then lo block
// are contiguous per ct so one TDM descriptor stages a whole column tile.
// ============================================================
__global__ void vq_pack_cb(const float* __restrict__ cb, uint32_t* __restrict__ pk) {
    int t    = blockIdx.x * 256 + threadIdx.x;   // 131072 threads
    int j    = t & 7;
    int lane = (t >> 3) & 31;
    int dt   = (t >> 8) & 7;
    int ct   = t >> 11;
    int n    = ct * 16 + (lane & 15);
    int lsel = lane >> 4;
    int k0   = (j < 4) ? (lsel * 8 + 2 * j) : (16 + lsel * 8 + 2 * (j - 4));
    int d0   = dt * 32 + k0;

    float v0 = cb[n * D_DIM + d0];
    float v1 = cb[n * D_DIM + d0 + 1];
    uint32_t h0 = bf16_rn_bits(v0), h1 = bf16_rn_bits(v1);
    uint32_t l0 = bf16_rn_bits(v0 - bf16_bits_to_f(h0));
    uint32_t l1 = bf16_rn_bits(v1 - bf16_bits_to_f(h1));

    int base = ct * 4096 + dt * 256 + lane * 8 + j;
    pk[base]        = h0 | (h1 << 16);   // hi half of the 16 KB block
    pk[base + 2048] = l0 | (l1 << 16);   // lo half
}

// ============================================================
// Kernel 2: per-codeword squared norms (||x||^2 is argmin-invariant).
// ============================================================
__global__ void vq_norms(const float* __restrict__ cb, float* __restrict__ nrm) {
    int k = blockIdx.x * 256 + threadIdx.x;
    if (k < K_CB) {
        float s = 0.f;
        for (int d = 0; d < D_DIM; ++d) { float v = cb[k * D_DIM + d]; s += v * v; }
        nrm[k] = s;
    }
}

// ============================================================
// Kernel 3: fused split-bf16 WMMA GEMM + streaming argmin.
// 256 blocks x 256 threads (8 waves). Wave owns 16 pixels; A (hi+lo) lives
// in VGPRs for the whole kernel. B column tiles are DMA'd by the Tensor
// Data Mover into a double-buffered LDS (TENSORcnt), overlapping the next
// tile's transfer with the current tile's 24 WMMAs.
// ============================================================
__global__ __launch_bounds__(256, 1)
void vq_argmin(const float* __restrict__ x,
               const uint32_t* __restrict__ cbpk,
               const float* __restrict__ nrm,
               int* __restrict__ idx_out, float* __restrict__ idx_f_out) {
    __shared__ alignas(16) uint16_t Ahs[8][16][32];   // per-wave A staging (hi)
    __shared__ alignas(16) uint16_t Als[8][16][32];   // per-wave A staging (lo)
    __shared__ alignas(16) uint32_t Bbuf[2][4096];    // 2 x 16 KB column tiles

    const int tid = threadIdx.x, wave = tid >> 5, lane = tid & 31;
    const int r = lane & 15, sel = lane >> 4;
    const int rowbase = blockIdx.x * 128 + wave * 16;

    // pixel n -> x offset: x[b, c, h, w] = b*262144 + c*1024 + (h*32+w)
    const int bb = rowbase >> 10, hw = rowbase & 1023;
    const float* xp = x + bb * 262144 + hw;

#if HAVE_TDM
    const uint32_t lds_b0 = (uint32_t)(uintptr_t)(void*)&Bbuf[0][0];
    const uint32_t lds_b1 = (uint32_t)(uintptr_t)(void*)&Bbuf[1][0];
    if (wave == 0) tdm_load_16kb(cbpk, lds_b0);       // prologue: tile 0 in flight
#endif

    // ---- stage A (once): 16 pixels x 256 dims, split bf16, swizzle via LDS ----
    U16bf ah[8], al[8];
    #pragma unroll
    for (int dt = 0; dt < 8; ++dt) {
        #pragma unroll
        for (int it = 0; it < 16; ++it) {
            int cl = 2 * it + sel;                       // 0..31 dim-in-tile
            float v = xp[(dt * 32 + cl) * 1024 + r];     // coalesced 64B per dim
            uint32_t h = bf16_rn_bits(v);
            uint32_t l = bf16_rn_bits(v - bf16_bits_to_f(h));
            Ahs[wave][r][cl] = (uint16_t)h;
            Als[wave][r][cl] = (uint16_t)l;
        }
        asm volatile("s_wait_dscnt 0" ::: "memory");
        const uint4* Ah4 = (const uint4*)&Ahs[wave][0][0];
        const uint4* Al4 = (const uint4*)&Als[wave][0][0];
        ah[dt].q[0] = Ah4[r * 4 + sel];       // K 0..7   / 8..15
        ah[dt].q[1] = Ah4[r * 4 + 2 + sel];   // K 16..23 / 24..31
        al[dt].q[0] = Al4[r * 4 + sel];
        al[dt].q[1] = Al4[r * 4 + 2 + sel];
        asm volatile("s_wait_dscnt 0" ::: "memory");
    }

    float minv[8]; int mini[8];
    #pragma unroll
    for (int j = 0; j < 8; ++j) { minv[j] = 3.4e38f; mini[j] = 0; }

    for (int ct = 0; ct < CT_TILES; ++ct) {
        __syncthreads();   // everyone done with buf[(ct-1)&1] before it is overwritten
#if HAVE_TDM
        if (wave == 0) {
            if (ct + 1 < CT_TILES) {
                // kick DMA of the next tile into the other buffer, then wait
                // only for the current tile (TDM ops complete in issue order)
                tdm_load_16kb(cbpk + (ct + 1) * 4096, (ct & 1) ? lds_b0 : lds_b1);
                __builtin_amdgcn_s_wait_tensorcnt(1);
            } else {
                __builtin_amdgcn_s_wait_tensorcnt(0);
            }
        }
#else
        // fallback: cooperative synchronous copy of the 16 KB tile
        const uint4* s4 = (const uint4*)(cbpk + ct * 4096);
        uint4* d4 = (uint4*)&Bbuf[ct & 1][0];
        d4[tid]       = s4[tid];
        d4[tid + 256] = s4[tid + 256];
        d4[tid + 512] = s4[tid + 512];
        d4[tid + 768] = s4[tid + 768];
        if (ct + 1 < CT_TILES) __builtin_prefetch(cbpk + (ct + 1) * 4096, 0, 1);
#endif
        __syncthreads();   // publish staged tile to all waves

        v8f acc = {};
        const uint4* Bh4 = (const uint4*)&Bbuf[ct & 1][0];
        const uint4* Bl4 = (const uint4*)&Bbuf[ct & 1][2048];
        #pragma unroll
        for (int dt = 0; dt < 8; ++dt) {
            U16bf bh, bl;
            bh.q[0] = Bh4[dt * 64 + lane * 2]; bh.q[1] = Bh4[dt * 64 + lane * 2 + 1];
            bl.q[0] = Bl4[dt * 64 + lane * 2]; bl.q[1] = Bl4[dt * 64 + lane * 2 + 1];
            // split-f32 product: hi*hi + hi*lo + lo*hi  (~f32-accurate dot)
            acc = __builtin_amdgcn_wmma_f32_16x16x32_bf16(false, ah[dt].v, false, bh.v, (short)0, acc, false, false);
            acc = __builtin_amdgcn_wmma_f32_16x16x32_bf16(false, ah[dt].v, false, bl.v, (short)0, acc, false, false);
            acc = __builtin_amdgcn_wmma_f32_16x16x32_bf16(false, al[dt].v, false, bh.v, (short)0, acc, false, false);
        }
        const int col = ct * 16 + r;
        const float cn = nrm[col];
        #pragma unroll
        for (int j = 0; j < 8; ++j) {
            float dist = cn - 2.0f * acc[j];   // + ||x||^2 omitted (row-constant)
            if (dist < minv[j]) { minv[j] = dist; mini[j] = col; }
        }
    }

    // reduce (min, idx) across the 16 lanes holding each row (wave32 halves)
    #pragma unroll
    for (int j = 0; j < 8; ++j) {
        float v = minv[j]; int i = mini[j];
        #pragma unroll
        for (int m = 8; m >= 1; m >>= 1) {
            float ov = __shfl_xor(v, m, 32);
            int   oi = __shfl_xor(i, m, 32);
            if (ov < v || (ov == v && oi < i)) { v = ov; i = oi; }
        }
        if (r == 0) {                               // lanes 0 and 16
            int row = rowbase + sel * 8 + j;        // C layout: M = j + 8*sel
            idx_out[row]   = i;
            idx_f_out[row] = (float)i;
        }
    }
}

// ============================================================
// Kernel 4: one-hot encodings (134 MB, pure store bandwidth).
// ============================================================
__global__ void vq_enc(const int* __restrict__ idx, float* __restrict__ enc) {
    int gid = blockIdx.x * 256 + threadIdx.x;   // 8,388,608 threads, 4 f32 each
    int n  = gid >> 8;
    int kb = (gid & 255) << 2;
    int id = idx[n];
    float* p = enc + (size_t)n * 1024 + kb;     // 4B-aligned region: scalar stores
    p[0] = (kb + 0 == id) ? 1.0f : 0.0f;
    p[1] = (kb + 1 == id) ? 1.0f : 0.0f;
    p[2] = (kb + 2 == id) ? 1.0f : 0.0f;
    p[3] = (kb + 3 == id) ? 1.0f : 0.0f;
}

// ============================================================
// Kernel 5: quantized output gather + per-block squared-error partials.
// ============================================================
__global__ void vq_quant(const float* __restrict__ x, const float* __restrict__ cb,
                         const int* __restrict__ idx,
                         float* __restrict__ outq, float* __restrict__ part) {
    __shared__ float red[256];
    int tid = threadIdx.x;
    int e = blockIdx.x * 256 + tid;        // 8,388,608 elements
    int bb = e >> 18;
    int c  = (e >> 10) & 255;
    int hw = e & 1023;
    int n  = (bb << 10) + hw;
    float q  = cb[idx[n] * 256 + c];       // codebook is L2-hot (1 MB)
    float xv = x[e];
    float d  = q - xv;
    outq[e] = q;                           // straight-through fwd value == q
    red[tid] = d * d;
    __syncthreads();
    for (int s = 128; s > 0; s >>= 1) {
        if (tid < s) red[tid] += red[tid + s];
        __syncthreads();
    }
    if (tid == 0) part[blockIdx.x] = red[0];
}

// ============================================================
// Kernel 6: deterministic fixed-order final loss reduction.
// loss = q_mse + 0.25*e_mse = 1.25 * mean((q-x)^2)
// ============================================================
__global__ void vq_loss(const float* __restrict__ part, float* __restrict__ loss) {
    __shared__ float red[256];
    int tid = threadIdx.x;
    float s = 0.f;
    for (int i = 0; i < 128; ++i) s += part[tid + 256 * i];
    red[tid] = s;
    __syncthreads();
    for (int t = 128; t > 0; t >>= 1) {
        if (tid < t) red[tid] += red[tid + t];
        __syncthreads();
    }
    if (tid == 0) loss[0] = red[0] * (1.25f / (float)N_ELEM);
}

// ============================================================
extern "C" void kernel_launch(void* const* d_in, const int* in_sizes, int n_in,
                              void* d_out, int out_size, void* d_ws, size_t ws_size,
                              hipStream_t stream) {
    const float* x  = (const float*)d_in[0];   // [32,256,32,32] f32
    const float* cb = (const float*)d_in[1];   // [1024,256] f32
    float* out = (float*)d_out;
    char* ws = (char*)d_ws;

    uint32_t* cbpk = (uint32_t*)(ws + WS_CBPK);
    float*    nrm  = (float*)(ws + WS_NRM);
    int*      idx  = (int*)(ws + WS_IDX);
    float*    part = (float*)(ws + WS_PART);

    vq_pack_cb<<<512, 256, 0, stream>>>(cb, cbpk);
    vq_norms <<<4, 256, 0, stream>>>(cb, nrm);
    vq_argmin<<<256, 256, 0, stream>>>(x, cbpk, nrm, idx, out + OUT_IDX);
    vq_enc   <<<32768, 256, 0, stream>>>(idx, out + OUT_ENC);
    vq_quant <<<32768, 256, 0, stream>>>(x, cb, idx, out + OUT_Q, part);
    vq_loss  <<<1, 256, 0, stream>>>(part, out);
}
